// Linear4bit_5214090297852
// MI455X (gfx1250) — 8-bit matrix operand order repaired and checked by reading, compiled, not run on hardware
//
#include <hip/hip_runtime.h>
#include <hip/hip_bf16.h>

typedef __attribute__((ext_vector_type(8))) int v8i;

#define DDIM 4096
#define OPT_SCALE 2.513930578568423f
#define HAD_SCALE 0.08838834764831845f

#if defined(__has_builtin)
#if __has_builtin(__builtin_amdgcn_global_load_async_to_lds_b128)
#define USE_ASYNC_LDS 1
#endif
#endif

typedef int v4i_vs __attribute__((vector_size(16)));
typedef __attribute__((address_space(1))) v4i_vs* gbl_v4_t;
typedef __attribute__((address_space(3))) v4i_vs* lds_v4_t;

#ifdef USE_ASYNC_LDS
#define ASYNC_B128(gp, lp, off) \
    __builtin_amdgcn_global_load_async_to_lds_b128((gbl_v4_t)(gp), (lds_v4_t)(lp), (off), 0)
#endif

__device__ __forceinline__ void wait_async_le(int tag4) {
#if defined(__has_builtin) && __has_builtin(__builtin_amdgcn_s_wait_asynccnt)
    if (tag4) __builtin_amdgcn_s_wait_asynccnt(4);
    else      __builtin_amdgcn_s_wait_asynccnt(0);
#else
    if (tag4) asm volatile("s_wait_asynccnt 0x4" ::: "memory");
    else      asm volatile("s_wait_asynccnt 0x0" ::: "memory");
#endif
}

__global__ __launch_bounds__(256) void hq4_kernel(const float* __restrict__ in,
                                                  signed char* __restrict__ q,
                                                  float* __restrict__ step_out) {
    __shared__ float s[DDIM];
    __shared__ float red[256];
    const int tid = threadIdx.x;
    const size_t row = blockIdx.x;
    const float* __restrict__ rp = in + row * DDIM;

#pragma unroll
    for (int i = 0; i < DDIM / 256; ++i) s[tid + 256 * i] = rp[tid + 256 * i];
    __syncthreads();

    for (int h = 1; h < 128; h <<= 1) {
        const int mask = h - 1;
#pragma unroll
        for (int it = 0; it < (DDIM / 2) / 256; ++it) {
            int t   = tid + 256 * it;
            int blk = t >> 6;
            int j   = t & 63;
            int i0  = (blk << 7) + ((j & ~mask) << 1) + (j & mask);
            int i1  = i0 + h;
            float a = s[i0], b = s[i1];
            s[i0] = a + b;
            s[i1] = a - b;
        }
        __syncthreads();
    }

    float ss = 0.f;
#pragma unroll
    for (int i = 0; i < DDIM / 256; ++i) {
        float v = s[tid + 256 * i] * HAD_SCALE;
        ss += v * v;
    }
    red[tid] = ss;
    __syncthreads();
    for (int off = 128; off > 0; off >>= 1) {
        if (tid < off) red[tid] += red[tid + off];
        __syncthreads();
    }
    const float stdv  = sqrtf(red[0] * (1.0f / DDIM)) + 1e-8f;
    const float scale = OPT_SCALE * stdv;
    const float step  = 2.0f * scale * (1.0f / 15.0f);
    const float istep = 1.0f / step;

    signed char* __restrict__ qp = q + row * DDIM;
#pragma unroll
    for (int i = 0; i < DDIM / 256; ++i) {
        int   idx = tid + 256 * i;
        float v   = s[idx] * HAD_SCALE;
        v = fminf(fmaxf(v, -scale), scale);
        int k = (int)rintf((v + scale) * istep);
        qp[idx] = (signed char)(2 * k - 15);
    }
    if (tid == 0) step_out[row] = step;
}

__global__ __launch_bounds__(256) void wmma_gemm_i8(
    const signed char* __restrict__ A,
    const signed char* __restrict__ B,
    const float* __restrict__ step_a,
    const float* __restrict__ step_b,
    const float* __restrict__ bias,
    float* __restrict__ out,
    int M, int N, int K) {
    __shared__ __align__(16) signed char As[2][128][80];
    __shared__ __align__(16) signed char Bs[2][128][80];

    const int tid  = threadIdx.x;
    const int lane = tid & 31;
    const int wave = tid >> 5;
    const int wr   = wave >> 2;
    const int wc   = wave & 3;
    const int m0   = blockIdx.y * 128;
    const int n0   = blockIdx.x * 128;

    const int lr = tid >> 1;
    const int lh = (tid & 1) * 32;
    const signed char* gA0 = A + (size_t)(m0 + lr) * K + lh;
    const signed char* gB0 = B + (size_t)(n0 + lr) * K + lh;

    v8i acc[4][2];
#pragma unroll
    for (int mi = 0; mi < 4; ++mi)
#pragma unroll
        for (int ni = 0; ni < 2; ++ni)
            acc[mi][ni] = (v8i){0, 0, 0, 0, 0, 0, 0, 0};

    const int frow = lane & 15;
    const int a_kb = (lane < 16) ? 0 : 16;
    const int b_kb = (lane < 16) ? 0 : 16;

#ifdef USE_ASYNC_LDS
    ASYNC_B128(gA0, &As[0][lr][lh], 0);
    ASYNC_B128(gA0, &As[0][lr][lh], 16);
    ASYNC_B128(gB0, &Bs[0][lr][lh], 0);
    ASYNC_B128(gB0, &Bs[0][lr][lh], 16);

    int buf = 0;
    for (int kk = 0; kk < K; kk += 64) {
        const bool has_next = (kk + 64) < K;
        __syncthreads();
        if (has_next) {
            const signed char* ga = gA0 + kk + 64;
            const signed char* gb = gB0 + kk + 64;
            ASYNC_B128(ga, &As[buf ^ 1][lr][lh], 0);
            ASYNC_B128(ga, &As[buf ^ 1][lr][lh], 16);
            ASYNC_B128(gb, &Bs[buf ^ 1][lr][lh], 0);
            ASYNC_B128(gb, &Bs[buf ^ 1][lr][lh], 16);
            wait_async_le(1);
        } else {
            wait_async_le(0);
        }
        __syncthreads();
#else
    int buf = 0;
    for (int kk = 0; kk < K; kk += 64) {
        const signed char* ap = gA0 + kk;
        const signed char* bp = gB0 + kk;
        int4 av0 = *(const int4*)(ap);
        int4 av1 = *(const int4*)(ap + 16);
        int4 bv0 = *(const int4*)(bp);
        int4 bv1 = *(const int4*)(bp + 16);
        __syncthreads();
        *(int4*)&As[0][lr][lh]      = av0;
        *(int4*)&As[0][lr][lh + 16] = av1;
        *(int4*)&Bs[0][lr][lh]      = bv0;
        *(int4*)&Bs[0][lr][lh + 16] = bv1;
        __syncthreads();
#endif
        v8i af[4], bf[2];
#pragma unroll
        for (int mi = 0; mi < 4; ++mi) {
            const signed char* rp = &As[buf][wr * 64 + mi * 16 + frow][0];
#pragma unroll
            for (int c = 0; c < 4; ++c) {
                int2 v = *(const int2*)(rp + a_kb + 8 * (c & 1) + 32 * (c >> 1));
                af[mi][2 * c]     = v.x;
                af[mi][2 * c + 1] = v.y;
            }
        }
#pragma unroll
        for (int ni = 0; ni < 2; ++ni) {
            const signed char* rp = &Bs[buf][wc * 32 + ni * 16 + frow][0];
            int4 lo = *(const int4*)(rp + b_kb);
            int4 hi = *(const int4*)(rp + b_kb + 32);
            bf[ni][0] = lo.x; bf[ni][1] = lo.y; bf[ni][2] = lo.z; bf[ni][3] = lo.w;
            bf[ni][4] = hi.x; bf[ni][5] = hi.y; bf[ni][6] = hi.z; bf[ni][7] = hi.w;
        }
#pragma unroll
        for (int mi = 0; mi < 4; ++mi)
#pragma unroll
            for (int ni = 0; ni < 2; ++ni)
                acc[mi][ni] = __builtin_amdgcn_wmma_i32_16x16x64_iu8(
                    true, af[mi], true, bf[ni], acc[mi][ni], false, false);
#ifdef USE_ASYNC_LDS
        buf ^= 1;
    }
#else
    }
#endif

    const int ncol  = lane & 15;
    const int mhalf = (lane < 16) ? 0 : 8;
#pragma unroll
    for (int ni = 0; ni < 2; ++ni) {
        int   col = n0 + wc * 32 + ni * 16 + ncol;
        float sb  = step_b[col] * 0.25f;
        float bv  = bias[col];
#pragma unroll
        for (int mi = 0; mi < 4; ++mi) {
#pragma unroll
            for (int e = 0; e < 8; ++e) {
                int   rowm = m0 + wr * 64 + mi * 16 + mhalf + e;
                float sa   = step_a[rowm];
                out[(size_t)rowm * N + col] = (float)acc[mi][ni][e] * sa * sb + bv;
            }
        }
    }
}

extern "C" void kernel_launch(void* const* d_in, const int* in_sizes, int n_in,
                              void* d_out, int out_size, void* d_ws, size_t ws_size,
                              hipStream_t stream) {
    const float* x    = (const float*)d_in[0];
    const float* w    = (const float*)d_in[1];
    const float* bias = (const float*)d_in[2];
    const int M = in_sizes[0] / DDIM;
    const int N = in_sizes[1] / DDIM;

    char* ws = (char*)d_ws;
    signed char* xq = (signed char*)ws;
    signed char* wq = xq + (size_t)M * DDIM;
    float* step_x   = (float*)(wq + (size_t)N * DDIM);
    float* step_w   = step_x + M;
    float* out      = (float*)d_out;

    hq4_kernel<<<M, 256, 0, stream>>>(x, xq, step_x);
    hq4_kernel<<<N, 256, 0, stream>>>(w, wq, step_w);

    dim3 grid(N / 128, M / 128);
    wmma_gemm_i8<<<grid, 256, 0, stream>>>(xq, wq, step_x, step_w, bias, out,
                                           M, N, DDIM);
}
